// Siamese_LSTM_41231686041635
// MI455X (gfx1250) — compile-verified, matrix operand order checked by reading, not run on hardware
//
#include <hip/hip_runtime.h>
#include <hip/hip_bf16.h>

// ---------------------------------------------------------------------------
// CDNA5 (gfx1250) Siamese-LSTM implementation.
// Persistent per-batch-tile LSTM blocks (h in LDS double buffer, c in VGPRs)
// using v_wmma_f32_16x16x32_bf16 for all gate GEMMs; bf16 activations between
// layers; WMMA FC head. __launch_bounds__(256,1) opens the full VGPR file.
// B-fragments for all 4 gates are batch-loaded before the WMMA block so the
// global loads clause together and overlap with WMMA issue.
// ---------------------------------------------------------------------------

typedef __attribute__((ext_vector_type(16))) __bf16 v16bf;
typedef __attribute__((ext_vector_type(8)))  float  v8f;

union FragU { int4 i4[2]; v16bf v; };

__device__ __forceinline__ unsigned short f2bf(float f) {
    unsigned u = __float_as_uint(f);
    unsigned r = u + 0x7FFFu + ((u >> 16) & 1u);   // round-to-nearest-even
    return (unsigned short)(r >> 16);
}
__device__ __forceinline__ float bf2f(unsigned short h) {
    return __uint_as_float(((unsigned)h) << 16);
}
__device__ __forceinline__ v8f zero_v8f() {
    v8f z = {0.f,0.f,0.f,0.f,0.f,0.f,0.f,0.f};
    return z;
}
__device__ __forceinline__ float sigmoid_f(float x) {
    return 1.0f / (1.0f + __expf(-x));
}
__device__ __forceinline__ float tanh_f(float x) {
    x = fminf(fmaxf(x, -15.f), 15.f);
    float e = __expf(-2.0f * x);
    return (1.0f - e) / (1.0f + e);
}

__device__ __forceinline__ v8f wmma_bf16(v16bf a, v16bf b, v8f c) {
    // v_wmma_f32_16x16x32_bf16: D = A(16x32) * B(32x16) + C(16x16 f32)
    return __builtin_amdgcn_wmma_f32_16x16x32_bf16(
        false, a, false, b, (short)0, c, false, false);
}

// Load a 16x32 bf16 fragment (A layout; also used for B from row-major [N,K]
// weights == B loaded as W^T column-major). Per ISA: lanes 0-15 hold K 0-7 &
// 16-23 of row `lane`; lanes 16-31 hold K 8-15 & 24-31 of row `lane-16`.
__device__ __forceinline__ v16bf load_frag(const unsigned short* base, int ldk, int kc) {
    const int lane = threadIdx.x & 31;
    const int ln = lane & 15;
    const int hi = lane >> 4;
    const unsigned short* p = base + (size_t)ln * ldk + kc;
    FragU u;
    u.i4[0] = *(const int4*)(p + (hi ? 8 : 0));
    u.i4[1] = *(const int4*)(p + (hi ? 24 : 16));
    return u.v;
}

// ---------------------------------------------------------------------------
// fp32 -> bf16 weight conversion with K-padding: src [R][C] -> dst [R][Cpad]
// ---------------------------------------------------------------------------
__global__ void f32_to_bf16_pad(const float* __restrict__ src,
                                unsigned short* __restrict__ dst,
                                int R, int C, int Cpad) {
    for (size_t idx = (size_t)blockIdx.x * blockDim.x + threadIdx.x;
         idx < (size_t)R * Cpad; idx += (size_t)gridDim.x * blockDim.x) {
        int r = (int)(idx / Cpad);
        int c = (int)(idx - (size_t)r * Cpad);
        dst[idx] = (c < C) ? f2bf(src[(size_t)r * C + c]) : (unsigned short)0;
    }
}

// ---------------------------------------------------------------------------
// Persistent LSTM layer. One block = 32 batch rows for all T timesteps.
//   gates[32, 4H] = [xt | h] @ [Wih | Whh]^T + b   (WMMA bf16, f32 accum)
// Wave w owns hidden-column tiles jt (jt % 8 == w) and computes the 4 gate
// tiles (i,f,g,o) for those columns itself, so the c/h update is in-register.
// h double-buffered in LDS (bf16); relu(h) streamed to [B,T,H] bf16 (+f32 opt).
// ---------------------------------------------------------------------------
template<int HIN, int KIN, int H, bool FIRST, int NJMAX>
__global__ void __launch_bounds__(256, 1) lstm_layer_kernel(
    const unsigned short* __restrict__ Wih,   // [4H][KIN] bf16
    const unsigned short* __restrict__ Whh,   // [4H][H]   bf16
    const float* __restrict__ bias,           // [4H]
    const float* __restrict__ x_f32,          // FIRST: [B, HIN, T] fp32
    const unsigned short* __restrict__ x_bf,  // !FIRST: [B, T, HIN] bf16
    unsigned short* __restrict__ out_bf,      // [B, T, H] relu(h) bf16
    float* __restrict__ out_f32,              // optional fp32 relu(h)
    int B, int T)
{
    extern __shared__ unsigned short smem[];
    unsigned short* xs = smem;                 // [32][KIN]
    unsigned short* hs = smem + 32 * KIN;      // [2][32][H]

    const int tid  = threadIdx.x;
    const int wave = tid >> 5;
    const int lane = tid & 31;
    const int ln   = lane & 15;
    const int hi   = lane >> 4;
    const int b0   = blockIdx.x * 32;
    constexpr int NJT = H / 16;

    for (int i = tid; i < 2 * 32 * H; i += blockDim.x) hs[i] = 0;

    float c_state[NJMAX][2][8];
#pragma unroll
    for (int a = 0; a < NJMAX; ++a)
#pragma unroll
        for (int m = 0; m < 2; ++m)
#pragma unroll
            for (int v = 0; v < 8; ++v) c_state[a][m][v] = 0.f;

    __syncthreads();

    int cur = 0;
    for (int t = 0; t < T; ++t) {
        // ---- stage xt tile [32][KIN] into LDS as bf16 ----
        if (FIRST) {
            for (int idx = tid; idx < 32 * KIN; idx += blockDim.x) {
                int r = idx / KIN, f = idx - r * KIN;
                float v = 0.f;
                if (f < HIN && (b0 + r) < B)
                    v = x_f32[((size_t)(b0 + r) * HIN + f) * T + t];
                xs[idx] = f2bf(v);
            }
        } else {
            for (int idx = tid; idx < 32 * (KIN / 8); idx += blockDim.x) {
                int r = idx / (KIN / 8), cb = idx - r * (KIN / 8);
                int4 v = {0, 0, 0, 0};
                if ((b0 + r) < B)
                    v = ((const int4*)(x_bf + ((size_t)(b0 + r) * T + t) * HIN))[cb];
                ((int4*)xs)[idx] = v;
            }
        }
        __syncthreads();

        const unsigned short* hcur = hs + cur * 32 * H;
        unsigned short*       hnxt = hs + (cur ^ 1) * 32 * H;

#pragma unroll
        for (int jti = 0; jti < NJMAX; ++jti) {
            const int jt = wave + jti * 8;
            if (jt < NJT) {                         // wave-uniform predicate
                const int jb = jt * 16;
                v8f acc[4][2];
#pragma unroll
                for (int g = 0; g < 4; ++g)
#pragma unroll
                    for (int m = 0; m < 2; ++m) acc[g][m] = zero_v8f();

                // input contribution: xt @ Wih^T
#pragma unroll 1
                for (int kc = 0; kc < KIN; kc += 32) {
                    v16bf bfr[4];
#pragma unroll
                    for (int g = 0; g < 4; ++g)
                        bfr[g] = load_frag(Wih + (size_t)(g * H + jb) * KIN, KIN, kc);
                    v16bf a0 = load_frag(xs, KIN, kc);
                    v16bf a1 = load_frag(xs + 16 * KIN, KIN, kc);
#pragma unroll
                    for (int g = 0; g < 4; ++g) {
                        acc[g][0] = wmma_bf16(a0, bfr[g], acc[g][0]);
                        acc[g][1] = wmma_bf16(a1, bfr[g], acc[g][1]);
                    }
                }
                // recurrent contribution: h @ Whh^T
#pragma unroll 1
                for (int kc = 0; kc < H; kc += 32) {
                    v16bf bfr[4];
#pragma unroll
                    for (int g = 0; g < 4; ++g)
                        bfr[g] = load_frag(Whh + (size_t)(g * H + jb) * H, H, kc);
                    v16bf a0 = load_frag(hcur, H, kc);
                    v16bf a1 = load_frag(hcur + 16 * H, H, kc);
#pragma unroll
                    for (int g = 0; g < 4; ++g) {
                        acc[g][0] = wmma_bf16(a0, bfr[g], acc[g][0]);
                        acc[g][1] = wmma_bf16(a1, bfr[g], acc[g][1]);
                    }
                }

                const float bi = bias[0 * H + jb + ln];
                const float bf_ = bias[1 * H + jb + ln];
                const float bg = bias[2 * H + jb + ln];
                const float bo = bias[3 * H + jb + ln];
#pragma unroll
                for (int m = 0; m < 2; ++m) {
#pragma unroll
                    for (int v = 0; v < 8; ++v) {
                        const int row = m * 16 + v + hi * 8;
                        const float ig = sigmoid_f(acc[0][m][v] + bi);
                        const float fg = sigmoid_f(acc[1][m][v] + bf_);
                        const float gg = tanh_f(acc[2][m][v] + bg);
                        const float og = sigmoid_f(acc[3][m][v] + bo);
                        const float c  = fg * c_state[jti][m][v] + ig * gg;
                        const float h  = og * tanh_f(c);
                        c_state[jti][m][v] = c;
                        hnxt[(size_t)row * H + jb + ln] = f2bf(h);
                        const float hr = h > 0.f ? h : 0.f;
                        if ((b0 + row) < B) {
                            size_t o = ((size_t)(b0 + row) * T + t) * H + jb + ln;
                            out_bf[o] = f2bf(hr);
                            if (out_f32) out_f32[o] = hr;
                        }
                    }
                }
            }
        }
        __syncthreads();
        cur ^= 1;
    }
}

// ---------------------------------------------------------------------------
// WMMA FC layer: out[B,N] = relu(A[B,K](bf16) @ W[N,K]^T + bias), bf16 out.
// A-tile [32][K] staged in LDS; each wave sweeps N-column tiles.
// ---------------------------------------------------------------------------
template<int K, int N>
__global__ void __launch_bounds__(256, 1) fc_relu_kernel(
    const unsigned short* __restrict__ A,
    const unsigned short* __restrict__ W,
    const float* __restrict__ bias,
    unsigned short* __restrict__ out, int B)
{
    extern __shared__ unsigned short smem[];
    unsigned short* as = smem;    // [32][K]
    const int tid = threadIdx.x, wave = tid >> 5, lane = tid & 31;
    const int ln = lane & 15, hi = lane >> 4;
    const int b0 = blockIdx.x * 32;

    for (int idx = tid; idx < 32 * (K / 8); idx += blockDim.x) {
        int r = idx / (K / 8), cb = idx - r * (K / 8);
        int4 v = {0, 0, 0, 0};
        if (b0 + r < B) v = ((const int4*)(A + (size_t)(b0 + r) * K))[cb];
        ((int4*)as)[idx] = v;
    }
    __syncthreads();

    for (int nt = wave; nt < N / 16; nt += 8) {
        v8f acc0 = zero_v8f(), acc1 = zero_v8f();
#pragma unroll 2
        for (int kc = 0; kc < K; kc += 32) {
            v16bf bf = load_frag(W + (size_t)nt * 16 * K, K, kc);
            v16bf a0 = load_frag(as, K, kc);
            v16bf a1 = load_frag(as + 16 * K, K, kc);
            acc0 = wmma_bf16(a0, bf, acc0);
            acc1 = wmma_bf16(a1, bf, acc1);
        }
        const float bb = bias[nt * 16 + ln];
#pragma unroll
        for (int v = 0; v < 8; ++v) {
            const int r0 = v + hi * 8;
            float x0 = acc0[v] + bb; x0 = x0 > 0.f ? x0 : 0.f;
            float x1 = acc1[v] + bb; x1 = x1 > 0.f ? x1 : 0.f;
            if (b0 + r0 < B)      out[(size_t)(b0 + r0)      * N + nt * 16 + ln] = f2bf(x0);
            if (b0 + 16 + r0 < B) out[(size_t)(b0 + 16 + r0) * N + nt * 16 + ln] = f2bf(x1);
        }
    }
}

// d[b, t*64+j] = |h3a - h3b|   (fp32 in, bf16 out for the head GEMM)
__global__ void absdiff_kernel(const float* __restrict__ a,
                               const float* __restrict__ b,
                               unsigned short* __restrict__ out, size_t n) {
    for (size_t i = (size_t)blockIdx.x * blockDim.x + threadIdx.x; i < n;
         i += (size_t)gridDim.x * blockDim.x)
        out[i] = f2bf(fabsf(a[i] - b[i]));
}

// final 16-wide dot: out[b] = d3[b,:] . Wf4 + bf4
__global__ void head_final_kernel(const unsigned short* __restrict__ d3,
                                  const float* __restrict__ Wf4,
                                  const float* __restrict__ bf4,
                                  float* __restrict__ out, int B) {
    int b = blockIdx.x * blockDim.x + threadIdx.x;
    if (b < B) {
        float s = bf4[0];
#pragma unroll
        for (int k = 0; k < 16; ++k) s += bf2f(d3[(size_t)b * 16 + k]) * Wf4[k];
        out[b] = s;
    }
}

// ---------------------------------------------------------------------------
static inline size_t align256(size_t x) { return (x + 255) & ~(size_t)255; }

extern "C" void kernel_launch(void* const* d_in, const int* in_sizes, int n_in,
                              void* d_out, int out_size, void* d_ws, size_t ws_size,
                              hipStream_t stream) {
    (void)n_in; (void)out_size; (void)ws_size;
    const float* x1    = (const float*)d_in[0];
    const float* x2    = (const float*)d_in[1];
    const float* W_ih1 = (const float*)d_in[2];
    const float* W_hh1 = (const float*)d_in[3];
    const float* b1    = (const float*)d_in[4];
    const float* W_ih2 = (const float*)d_in[5];
    const float* W_hh2 = (const float*)d_in[6];
    const float* b2    = (const float*)d_in[7];
    const float* W_ih3 = (const float*)d_in[8];
    const float* W_hh3 = (const float*)d_in[9];
    const float* b3    = (const float*)d_in[10];
    const float* Wf1   = (const float*)d_in[11];
    const float* bf1   = (const float*)d_in[12];
    const float* Wf2   = (const float*)d_in[13];
    const float* bf2   = (const float*)d_in[14];
    const float* Wf3   = (const float*)d_in[15];
    const float* bf3   = (const float*)d_in[16];
    const float* Wf4   = (const float*)d_in[17];
    const float* bf4   = (const float*)d_in[18];

    const int F = 58, T = 30;
    const int H1 = 512, H2 = 128, H3 = 64;
    const int B = in_sizes[0] / (F * T);

    // workspace layout
    char* p = (char*)d_ws;
    auto alloc = [&](size_t bytes) -> void* { void* r = (void*)p; p += align256(bytes); return r; };
    unsigned short* wih1b = (unsigned short*)alloc((size_t)4 * H1 * 64 * 2);
    unsigned short* whh1b = (unsigned short*)alloc((size_t)4 * H1 * H1 * 2);
    unsigned short* wih2b = (unsigned short*)alloc((size_t)4 * H2 * H1 * 2);
    unsigned short* whh2b = (unsigned short*)alloc((size_t)4 * H2 * H2 * 2);
    unsigned short* wih3b = (unsigned short*)alloc((size_t)4 * H3 * H2 * 2);
    unsigned short* whh3b = (unsigned short*)alloc((size_t)4 * H3 * H3 * 2);
    unsigned short* wf1b  = (unsigned short*)alloc((size_t)960 * 1920 * 2);
    unsigned short* wf2b  = (unsigned short*)alloc((size_t)480 * 960 * 2);
    unsigned short* wf3b  = (unsigned short*)alloc((size_t)16 * 480 * 2);
    unsigned short* h1buf = (unsigned short*)alloc((size_t)B * T * H1 * 2);
    unsigned short* h2buf = (unsigned short*)alloc((size_t)B * T * H2 * 2);
    float*          h3a   = (float*)alloc((size_t)B * T * H3 * 4);
    float*          h3b   = (float*)alloc((size_t)B * T * H3 * 4);
    unsigned short* dbuf  = (unsigned short*)alloc((size_t)B * 1920 * 2);  // also layer3 bf16 sink
    unsigned short* f1out = (unsigned short*)alloc((size_t)B * 960 * 2);
    unsigned short* f2out = (unsigned short*)alloc((size_t)B * 480 * 2);
    unsigned short* f3out = (unsigned short*)alloc((size_t)B * 16 * 2);

    // ---- convert weights to bf16 (K padded to multiples of 32) ----
    auto conv = [&](const float* s, unsigned short* d, int R, int C, int Cpad) {
        int n = R * Cpad;
        f32_to_bf16_pad<<<(n + 255) / 256, 256, 0, stream>>>(s, d, R, C, Cpad);
    };
    conv(W_ih1, wih1b, 4 * H1, F, 64);
    conv(W_hh1, whh1b, 4 * H1, H1, H1);
    conv(W_ih2, wih2b, 4 * H2, H1, H1);
    conv(W_hh2, whh2b, 4 * H2, H2, H2);
    conv(W_ih3, wih3b, 4 * H3, H2, H2);
    conv(W_hh3, whh3b, 4 * H3, H3, H3);
    conv(Wf1, wf1b, 960, 1920, 1920);
    conv(Wf2, wf2b, 480, 960, 960);
    conv(Wf3, wf3b, 16, 480, 480);

    const int nb = (B + 31) / 32;
    const size_t sm1 = (size_t)(32 * 64 + 2 * 32 * H1) * 2;   // 68 KB
    const size_t sm2 = (size_t)(32 * H1 + 2 * 32 * H2) * 2;   // 48 KB
    const size_t sm3 = (size_t)(32 * H2 + 2 * 32 * H3) * 2;   // 16 KB

    // ---- branch 1 (x1) ----
    lstm_layer_kernel<58, 64, 512, true, 4><<<nb, 256, sm1, stream>>>(
        wih1b, whh1b, b1, x1, nullptr, h1buf, nullptr, B, T);
    lstm_layer_kernel<512, 512, 128, false, 1><<<nb, 256, sm2, stream>>>(
        wih2b, whh2b, b2, nullptr, h1buf, h2buf, nullptr, B, T);
    lstm_layer_kernel<128, 128, 64, false, 1><<<nb, 256, sm3, stream>>>(
        wih3b, whh3b, b3, nullptr, h2buf, dbuf, h3a, B, T);

    // ---- branch 2 (x2), reusing the big activation buffers ----
    lstm_layer_kernel<58, 64, 512, true, 4><<<nb, 256, sm1, stream>>>(
        wih1b, whh1b, b1, x2, nullptr, h1buf, nullptr, B, T);
    lstm_layer_kernel<512, 512, 128, false, 1><<<nb, 256, sm2, stream>>>(
        wih2b, whh2b, b2, nullptr, h1buf, h2buf, nullptr, B, T);
    lstm_layer_kernel<128, 128, 64, false, 1><<<nb, 256, sm3, stream>>>(
        wih3b, whh3b, b3, nullptr, h2buf, dbuf, h3b, B, T);

    // ---- |h1 - h2| -> [B, 1920] bf16 ----
    {
        size_t n = (size_t)B * T * H3;
        absdiff_kernel<<<(int)((n + 255) / 256), 256, 0, stream>>>(h3a, h3b, dbuf, n);
    }

    // ---- FC head ----
    fc_relu_kernel<1920, 960><<<nb, 256, (size_t)32 * 1920 * 2, stream>>>(dbuf, wf1b, bf1, f1out, B);
    fc_relu_kernel<960, 480><<<nb, 256, (size_t)32 * 960 * 2, stream>>>(f1out, wf2b, bf2, f2out, B);
    fc_relu_kernel<480, 16><<<nb, 256, (size_t)32 * 480 * 2, stream>>>(f2out, wf3b, bf3, f3out, B);
    head_final_kernel<<<(B + 255) / 256, 256, 0, stream>>>(f3out, Wf4, bf4, (float*)d_out, B);
}
